// Hub_76132590289126
// MI455X (gfx1250) — matrix operand order repaired and checked by reading, not compiled, not run on hardware
//
#include <hip/hip_runtime.h>
#include <hip/hip_bf16.h>
#include <stdint.h>

// ---------------------------------------------------------------------------
// MI455X / gfx1250 (CDNA5, wave32).
// Pipeline: bf16 convert/transpose -> proj+bias+L2norm (WMMA) -> logits GEMM
//           (WMMA, 64KB async-LDS A staging, 2x2 tiles/wave) -> row softmax ->
//           weighted-sum GEMM (WMMA, double-buffered async-LDS staging)
//           fused with broadcast epilogue.
// All fragment loads are contiguous b128/b256 loads bit_cast into v16bf.
// ---------------------------------------------------------------------------

typedef __attribute__((ext_vector_type(16))) __bf16        v16bf;
typedef __attribute__((ext_vector_type(8)))  float         v8f;
typedef __attribute__((ext_vector_type(4)))  unsigned int  v4u;
typedef __attribute__((ext_vector_type(8)))  unsigned int  v8u;

#define DEVINL __device__ __forceinline__

struct U32x8 { v4u lo; v4u hi; };

// ---- bf16 conversion (RTNE, integer ops) ----------------------------------
DEVINL unsigned int f2bf_bits(float f) {
  unsigned int u = __builtin_bit_cast(unsigned int, f);
  u += 0x7FFFu + ((u >> 16) & 1u);
  return u >> 16;
}

// ---- WMMA fragment loaders (contiguous k = 16*half + s in both A and B, wave32) -------------------
// A (16x32 bf16 MxK): lane half*16+row holds K run [half*16,+16)
// -> two contiguous 16B loads (global or LDS).
DEVINL v16bf load_a_bf16(const unsigned short* base, int lda) {
  int lane = threadIdx.x & 31;
  int row = lane & 15, half = lane >> 4;
  const unsigned short* p = base + (size_t)row * lda + half * 16;
  U32x8 t;
  t.lo = *(const v4u*)(p);
  t.hi = *(const v4u*)(p + 8);
  return __builtin_bit_cast(v16bf, t);
}

// B (32x16 bf16 KxN) from [N][K] row-major (K-pairs contiguous, stride ldn):
// lane's 16 bf16 are one contiguous 32B run at col*ldn + half*16.
DEVINL v16bf load_b_kpair_bf16(const unsigned short* base, int ldn) {
  int lane = threadIdx.x & 31;
  int col = lane & 15, half = lane >> 4;
  v8u u = *(const v8u*)(base + (size_t)col * ldn + half * 16);
  return __builtin_bit_cast(v16bf, u);
}

DEVINL v8f wmma_bf16(v16bf a, v16bf b, v8f c) {
  return __builtin_amdgcn_wmma_f32_16x16x32_bf16(
      false, a, false, b, (short)0, c, false, false);
}

// ---- CDNA5 async global->LDS staging (GVS mode) ---------------------------
DEVINL void async_ld_lds_b128(unsigned lds_byte_addr, unsigned gbl_byte_off,
                              const void* sbase) {
  asm volatile("global_load_async_to_lds_b128 %0, %1, %2"
               :: "v"(lds_byte_addr), "v"(gbl_byte_off), "s"(sbase)
               : "memory");
}
DEVINL void wait_async0() {
  asm volatile("s_wait_asynccnt 0x0" ::: "memory");
}

// ---------------------------------------------------------------------------
// k0: f32 -> bf16 conversion, 4 elems/thread-iter (b128 in, b64 out).
// ---------------------------------------------------------------------------
__global__ void k0_convert(const float4* __restrict__ in,
                           uint2* __restrict__ outp, int n4) {
  int i = blockIdx.x * blockDim.x + threadIdx.x;
  int stride = gridDim.x * blockDim.x;
  for (; i < n4; i += stride) {
    float4 v = in[i];
    uint2 o;
    o.x = f2bf_bits(v.x) | (f2bf_bits(v.y) << 16);
    o.y = f2bf_bits(v.z) | (f2bf_bits(v.w) << 16);
    outp[i] = o;
  }
}

// ---------------------------------------------------------------------------
// k0t: memory [M=4096][H=1024] f32 -> memT_bf [H][M] bf16 (LDS-tiled).
// ---------------------------------------------------------------------------
__global__ __launch_bounds__(256) void k0_transpose(
    const float* __restrict__ in, unsigned short* __restrict__ outp) {
  __shared__ unsigned short tile[32][33];
  int bh = blockIdx.x & 31;   // H tile  (H/32 = 32)
  int bj = blockIdx.x >> 5;   // M tile  (M/32 = 128)
  int tx = threadIdx.x & 31, ty = threadIdx.x >> 5;  // 32 x 8
#pragma unroll
  for (int r = 0; r < 32; r += 8)
    tile[ty + r][tx] =
        (unsigned short)f2bf_bits(in[(size_t)(bj * 32 + ty + r) * 1024 + bh * 32 + tx]);
  __syncthreads();
#pragma unroll
  for (int r = 0; r < 32; r += 8)
    outp[(size_t)(bh * 32 + ty + r) * 4096 + bj * 32 + tx] = tile[tx][ty + r];
}

// ---------------------------------------------------------------------------
// k1: per 16-row strip: ta = a@Wa^T+ba, tb = b@Wb^T+bb fully in registers,
// row L2-norms via shfl_xor (ds_bpermute) + LDS ds_add_f32, emit
// c_bf = normalize(ta)+normalize(tb)  (bf16).
// ---------------------------------------------------------------------------
__global__ __launch_bounds__(256) void k1_proj_norm(
    const unsigned short* __restrict__ abf, const unsigned short* __restrict__ bbf,
    const unsigned short* __restrict__ wabf, const unsigned short* __restrict__ wbbf,
    const float* __restrict__ ba, const float* __restrict__ bb,
    unsigned short* __restrict__ cbf) {
  const int H = 1024;
  int strip = blockIdx.x;
  int wave = threadIdx.x >> 5;
  int lane = threadIdx.x & 31;
  int col = lane & 15, half = lane >> 4;

  __shared__ float ss[2][16];
  __shared__ float inv[2][16];
  if (threadIdx.x < 32) ss[threadIdx.x >> 4][threadIdx.x & 15] = 0.0f;
  __syncthreads();

  const unsigned short* X[2] = {abf, bbf};
  const unsigned short* W[2] = {wabf, wbbf};
  const float* BIAS[2] = {ba, bb};

  v8f acc[2][8];

#pragma unroll
  for (int m = 0; m < 2; ++m) {
    const unsigned short* abase = X[m] + (size_t)strip * 16 * H;
#pragma unroll
    for (int t = 0; t < 8; ++t) {
      int n0 = (wave + t * 8) * 16;
      const unsigned short* wbase = W[m] + (size_t)n0 * H;
      v8f c = {};
#pragma unroll 4
      for (int kk = 0; kk < H; kk += 32) {
        v16bf fa = load_a_bf16(abase + kk, H);
        v16bf fb = load_b_kpair_bf16(wbase + kk, H);
        c = wmma_bf16(fa, fb, c);
      }
      float bv = BIAS[m][n0 + col];
#pragma unroll
      for (int v = 0; v < 8; ++v) c[v] += bv;
      acc[m][t] = c;
#pragma unroll
      for (int v = 0; v < 8; ++v) {
        float p = c[v] * c[v];
        p += __shfl_xor(p, 1, 32);
        p += __shfl_xor(p, 2, 32);
        p += __shfl_xor(p, 4, 32);
        p += __shfl_xor(p, 8, 32);
        if (col == 0) atomicAdd(&ss[m][v + 8 * half], p);
      }
    }
  }
  __syncthreads();
  if (threadIdx.x < 32) {
    int m = threadIdx.x >> 4, r = threadIdx.x & 15;
    float n = sqrtf(ss[m][r]);
    inv[m][r] = 1.0f / fmaxf(n, 1e-12f);
  }
  __syncthreads();

#pragma unroll
  for (int t = 0; t < 8; ++t) {
    int n0 = (wave + t * 8) * 16;
#pragma unroll
    for (int v = 0; v < 8; ++v) {
      int row = v + 8 * half;
      float cv = acc[0][t][v] * inv[0][row] + acc[1][t][v] * inv[1][row];
      cbf[(size_t)(strip * 16 + row) * H + n0 + col] = (unsigned short)f2bf_bits(cv);
    }
  }
}

// ---------------------------------------------------------------------------
// k2: logits = c_bf @ mem_bf^T  [4096x4096] f32.
// Block = 8 waves, i-block of 32 rows (two 16-row A strips async-staged into
// 64KB LDS), j-block of 256 (wave owns 2 j-tiles). Each wave computes a 2x2
// grid of 16x16 tiles -> 4 WMMAs/k-step with full A/B fragment reuse.
// ---------------------------------------------------------------------------
__global__ __launch_bounds__(256) void k2_logits(
    const unsigned short* __restrict__ cbf,
    const unsigned short* __restrict__ membf,
    float* __restrict__ logits) {
  const int H = 1024, M = 4096;
  int wave = threadIdx.x >> 5;
  int lane = threadIdx.x & 31;
  int col = lane & 15, half = lane >> 4;
  int iblk = blockIdx.x >> 4;                      // 0..127 (32 rows each)
  int j0 = (blockIdx.x & 15) * 256 + wave * 32;    // 0..4064

  __shared__ alignas(16) unsigned short As[32 * 1024];  // 64 KB, two strips

  {
    unsigned ldsbase = (unsigned)(uintptr_t)(&As[0]);
    const unsigned short* gbase = cbf + (size_t)iblk * 32 * H;  // contiguous 64KB
#pragma unroll
    for (int r = 0; r < 16; ++r) {
      unsigned off = (unsigned)(threadIdx.x + r * 256) * 16u;
      async_ld_lds_b128(ldsbase + off, off, gbase);
    }
    wait_async0();
    __syncthreads();
  }

  const unsigned short* b0 = membf + (size_t)j0 * H;
  const unsigned short* b1 = membf + (size_t)(j0 + 16) * H;

  v8f c00 = {}, c01 = {}, c10 = {}, c11 = {};
#pragma unroll 2
  for (int kk = 0; kk < H; kk += 32) {
    v16bf fa0 = load_a_bf16(As + kk, H);
    v16bf fa1 = load_a_bf16(As + 16 * H + kk, H);
    v16bf f0 = load_b_kpair_bf16(b0 + kk, H);
    v16bf f1 = load_b_kpair_bf16(b1 + kk, H);
    c00 = wmma_bf16(fa0, f0, c00);
    c01 = wmma_bf16(fa0, f1, c01);
    c10 = wmma_bf16(fa1, f0, c10);
    c11 = wmma_bf16(fa1, f1, c11);
  }
  int row0 = iblk * 32 + 8 * half;
#pragma unroll
  for (int v = 0; v < 8; ++v) {
    logits[(size_t)(row0 + v) * M + j0 + col]            = c00[v];
    logits[(size_t)(row0 + v) * M + j0 + 16 + col]       = c01[v];
    logits[(size_t)(row0 + 16 + v) * M + j0 + col]       = c10[v];
    logits[(size_t)(row0 + 16 + v) * M + j0 + 16 + col]  = c11[v];
  }
}

// ---------------------------------------------------------------------------
// k3: row softmax over M=4096; row in registers (float4 IO), bf16 weights out.
// ---------------------------------------------------------------------------
__global__ __launch_bounds__(256) void k3_softmax(
    const float* __restrict__ logits, unsigned short* __restrict__ wbf) {
  const int M = 4096;
  int row = blockIdx.x;
  int tid = threadIdx.x;
  const float4* lr4 = (const float4*)(logits + (size_t)row * M);

  float4 lv[4];
  float mx = -3.4e38f;
#pragma unroll
  for (int cidx = 0; cidx < 4; ++cidx) {
    lv[cidx] = lr4[tid + cidx * 256];
    mx = fmaxf(mx, fmaxf(fmaxf(lv[cidx].x, lv[cidx].y), fmaxf(lv[cidx].z, lv[cidx].w)));
  }
  __shared__ float red[256];
  red[tid] = mx;
  __syncthreads();
  for (int s = 128; s > 0; s >>= 1) {
    if (tid < s) red[tid] = fmaxf(red[tid], red[tid + s]);
    __syncthreads();
  }
  mx = red[0];
  __syncthreads();

  float sum = 0.0f;
#pragma unroll
  for (int cidx = 0; cidx < 4; ++cidx) {
    lv[cidx].x = __expf(lv[cidx].x - mx);
    lv[cidx].y = __expf(lv[cidx].y - mx);
    lv[cidx].z = __expf(lv[cidx].z - mx);
    lv[cidx].w = __expf(lv[cidx].w - mx);
    sum += lv[cidx].x + lv[cidx].y + lv[cidx].z + lv[cidx].w;
  }
  red[tid] = sum;
  __syncthreads();
  for (int s = 128; s > 0; s >>= 1) {
    if (tid < s) red[tid] += red[tid + s];
    __syncthreads();
  }
  float rinv = 1.0f / red[0];

  uint2* w2 = (uint2*)(wbf + (size_t)row * M);
#pragma unroll
  for (int cidx = 0; cidx < 4; ++cidx) {
    uint2 o;
    o.x = f2bf_bits(lv[cidx].x * rinv) | (f2bf_bits(lv[cidx].y * rinv) << 16);
    o.y = f2bf_bits(lv[cidx].z * rinv) | (f2bf_bits(lv[cidx].w * rinv) << 16);
    w2[tid + cidx * 256] = o;
  }
}

// ---------------------------------------------------------------------------
// k4 staging: 32 rows x 512 cols bf16 of the weight strip (row stride M)
// into one 32KB LDS buffer via async DMA. 8 b128 issues/thread.
// ---------------------------------------------------------------------------
DEVINL void k4_stage(unsigned ldsbase, const unsigned short* strip, int jc) {
  const int M = 4096;
#pragma unroll
  for (int r = 0; r < 8; ++r) {
    int e = threadIdx.x + r * 256;   // 0..2047 16-byte chunks
    int rrow = e >> 6;               // 64 chunks per 512-col row
    int cw = e & 63;
    async_ld_lds_b128(ldsbase + (unsigned)e * 16u,
                      (unsigned)((rrow * M + jc + cw * 8) * 2), strip);
  }
}

// ---------------------------------------------------------------------------
// k4: wm = w_bf @ memory, fused with out[b,k,h] = wm[b,h]*Wc[k] + bc[k].
// i-block 32 rows, h-block 128 (8 waves x 16). Weight strip streamed in
// 512-col chunks through DOUBLE-BUFFERED async-LDS staging so the async DMA
// overlaps the WMMA pipe. B fragments from bf16 transposed memory (contiguous
// K-pairs), reused across both i-tiles. Output (134MB) written directly.
// ---------------------------------------------------------------------------
__global__ __launch_bounds__(256) void k4_wm_out(
    const unsigned short* __restrict__ wbf,
    const unsigned short* __restrict__ memTbf,   // [H][M]
    const float* __restrict__ Wc, const float* __restrict__ bc,
    float* __restrict__ out) {
  const int H = 1024, M = 4096, JC = 512;
  int wave = threadIdx.x >> 5;
  int lane = threadIdx.x & 31;
  int col = lane & 15, half = lane >> 4;
  int iblk = blockIdx.x >> 3;                       // 0..127 (32 rows)
  int h0 = ((blockIdx.x & 7) * 8 + wave) * 16;      // 0..1008

  __shared__ alignas(16) unsigned short Ws[2][32 * JC];  // 2 x 32 KB
  const unsigned short* strip = wbf + (size_t)iblk * 32 * M;
  const unsigned short* bbase = memTbf + (size_t)h0 * M;
  unsigned lds0 = (unsigned)(uintptr_t)(&Ws[0][0]);
  unsigned lds1 = (unsigned)(uintptr_t)(&Ws[1][0]);

  k4_stage(lds0, strip, 0);

  v8f c0 = {}, c1 = {};
  int buf = 0;
  for (int jc = 0; jc < M; jc += JC, buf ^= 1) {
    wait_async0();
    __syncthreads();
    if (jc + JC < M) k4_stage(buf ? lds0 : lds1, strip, jc + JC);

    const unsigned short* cur = &Ws[buf][0];
#pragma unroll 2
    for (int kk = 0; kk < JC; kk += 32) {
      v16bf fb  = load_b_kpair_bf16(bbase + jc + kk, M);
      v16bf fa0 = load_a_bf16(cur + kk, JC);
      v16bf fa1 = load_a_bf16(cur + 16 * JC + kk, JC);
      c0 = wmma_bf16(fa0, fb, c0);
      c1 = wmma_bf16(fa1, fb, c1);
    }
    __syncthreads();
  }

  float wc[8], bcv[8];
#pragma unroll
  for (int k = 0; k < 8; ++k) { wc[k] = Wc[k]; bcv[k] = bc[k]; }
#pragma unroll
  for (int v = 0; v < 8; ++v) {
    int row = iblk * 32 + v + 8 * half;
    float w0 = c0[v], w1 = c1[v];
#pragma unroll
    for (int k = 0; k < 8; ++k) {
      out[((size_t)row * 8 + k) * H + h0 + col]        = w0 * wc[k] + bcv[k];
      out[((size_t)(row + 16) * 8 + k) * H + h0 + col] = w1 * wc[k] + bcv[k];
    }
  }
}

// ---------------------------------------------------------------------------
// Workspace layout (bytes):
//   [0,         8388608)    mem_bf   (4096*1024 bf16)
//   [8388608,   16777216)   memT_bf  (1024*4096 bf16)
//   [16777216,  18874368)   Wa_bf
//   [18874368,  20971520)   Wb_bf
//   [20971520,  29360128)   c_bf
//   [29360128,  96468992)   logits f32  (first 16.8MB aliased as a_bf/b_bf
//                                        during stage 1, before k2 runs)
//   [96468992,  130023424)  w_bf
// ---------------------------------------------------------------------------
extern "C" void kernel_launch(void* const* d_in, const int* in_sizes, int n_in,
                              void* d_out, int out_size, void* d_ws,
                              size_t ws_size, hipStream_t stream) {
  const float* a   = (const float*)d_in[0];
  const float* b   = (const float*)d_in[1];
  const float* mem = (const float*)d_in[2];
  const float* Wa  = (const float*)d_in[3];
  const float* ba  = (const float*)d_in[4];
  const float* Wb  = (const float*)d_in[5];
  const float* bb  = (const float*)d_in[6];
  const float* Wc  = (const float*)d_in[7];
  const float* bc  = (const float*)d_in[8];
  float* out = (float*)d_out;

  char* ws = (char*)d_ws;
  unsigned short* membf  = (unsigned short*)(ws);
  unsigned short* memTbf = (unsigned short*)(ws + 8388608);
  unsigned short* wabf   = (unsigned short*)(ws + 16777216);
  unsigned short* wbbf   = (unsigned short*)(ws + 18874368);
  unsigned short* cbf    = (unsigned short*)(ws + 20971520);
  float*          logit  = (float*)(ws + 29360128);
  unsigned short* wbf    = (unsigned short*)(ws + 96468992);
  // stage-1 aliases inside the logits region (dead once k2 writes logits)
  unsigned short* abf    = (unsigned short*)(ws + 29360128);
  unsigned short* bbf    = (unsigned short*)(ws + 29360128 + 8388608);

  k0_convert<<<1024, 256, 0, stream>>>((const float4*)a,   (uint2*)abf,   1048576);
  k0_convert<<<1024, 256, 0, stream>>>((const float4*)b,   (uint2*)bbf,   1048576);
  k0_convert<<<1024, 256, 0, stream>>>((const float4*)mem, (uint2*)membf, 1048576);
  k0_convert<<<512,  256, 0, stream>>>((const float4*)Wa,  (uint2*)wabf,  262144);
  k0_convert<<<512,  256, 0, stream>>>((const float4*)Wb,  (uint2*)wbbf,  262144);
  k0_transpose<<<4096, 256, 0, stream>>>(mem, memTbf);
  k1_proj_norm<<<256, 256, 0, stream>>>(abf, bbf, wabf, wbbf, ba, bb, cbf);
  k2_logits<<<2048, 256, 0, stream>>>(cbf, membf, logit);
  k3_softmax<<<4096, 256, 0, stream>>>(logit, wbf);
  k4_wm_out<<<1024, 256, 0, stream>>>(wbf, memTbf, Wc, bc, out);
}
